// MultiHeadAttention_75471165325765
// MI455X (gfx1250) — matrix operand order repaired and checked by reading, not compiled, not run on hardware
//
#include <hip/hip_runtime.h>
#include <cstdint>
#include <cstddef>

typedef __attribute__((ext_vector_type(16))) __bf16 v16bf;
typedef __attribute__((ext_vector_type(8)))  __bf16 v8bf;
typedef __attribute__((ext_vector_type(8)))  float  v8f;

#define BB 2
#define SS 2048
#define DIM 1024
#define HH 16
#define DEPTH 64
#define SCALE 0.125f   // 1/sqrt(64)

// ---------- helpers ----------

__device__ __forceinline__ __bf16 f2bf(float f) {
  unsigned u = __builtin_bit_cast(unsigned, f);
  u = (u + 0x7FFFu + ((u >> 16) & 1u)) >> 16;   // round-to-nearest-even
  unsigned short h = (unsigned short)u;
  return __builtin_bit_cast(__bf16, h);
}

__device__ __forceinline__ v8f wmma_bf16(v16bf a, v16bf b, v8f c) {
  // D = A(16x32 bf16) * B(32x16 bf16) + C(16x16 f32)
  return __builtin_amdgcn_wmma_f32_16x16x32_bf16(false, a, false, b, (short)0, c,
                                                 false, false);
}

// Async global->LDS copy, 16B per lane. VDST = LDS byte offset (low 32 bits of
// a generic pointer to __shared__), VADDR = 64-bit global address, SADDR=off.
// Tracked with ASYNCcnt (CDNA5 ISA).
__device__ __forceinline__ void async_ldg_b128(unsigned lds_off, const void* g) {
  asm volatile("global_load_async_to_lds_b128 %0, %1, off"
               :: "v"(lds_off), "v"(g)
               : "memory");
}
__device__ __forceinline__ void wait_async0() {
  asm volatile("s_wait_asynccnt 0x0" ::: "memory");
}

// A fragment 16x32 (MxK), source row-major [m][k] starting at `base`, stride lda.
// ISA layout: lanes 0-15 -> M=lane, K = {0..7, 16..23}; lanes 16-31 -> K = {8..15, 24..31}.
__device__ __forceinline__ v16bf load_a_frag(const __bf16* base, int lda) {
  const int lane = threadIdx.x & 31;
  const int hf = lane >> 4, r = lane & 15;
  const __bf16* p = base + r * lda + hf * 8;
  v8bf lo = *(const v8bf*)p;          // K = hf*8 .. +7
  v8bf hi = *(const v8bf*)(p + 16);   // K = 16 + hf*8 .. +7
  v16bf a;
#pragma unroll
  for (int i = 0; i < 8; ++i) { a[i] = lo[i]; a[8 + i] = hi[i]; }
  return a;
}

// B fragment 32x16 (KxN) where B[k,n] = SRC[n][k]; SRC row-major over n, stride ldb.
// ISA layout: lanes 0-15 -> N=lane, K = {0..7, 16..23}; lanes 16-31 -> K = {8..15, 24..31}.
__device__ __forceinline__ v16bf load_b_frag(const __bf16* base, int ldb) {
  const int lane = threadIdx.x & 31;
  const int kh = lane >> 4, n = lane & 15;
  const __bf16* p = base + n * ldb + kh * 8;
  v8bf lo = *(const v8bf*)p;
  v8bf hi = *(const v8bf*)(p + 16);
  v16bf b;
#pragma unroll
  for (int i = 0; i < 8; ++i) { b[i] = lo[i]; b[8 + i] = hi[i]; }
  return b;
}

__device__ __forceinline__ v8f vzero8() {
  v8f z = {0.f, 0.f, 0.f, 0.f, 0.f, 0.f, 0.f, 0.f};
  return z;
}

// ---------- GEMM: Y[m,n] = sum_k A[m,k]*W[n,k] + bias[n] ----------
// MODE 0: A is f32 (converted on the fly), store bf16 head-split [B,H,S,DEPTH]
// MODE 1: A is bf16 (ctx, staged via async-to-LDS), store f32 row-major [M,DIM]
template <int MODE>
__global__ __launch_bounds__(256) void gemm_kernel(
    const float* __restrict__ Af, const __bf16* __restrict__ Ab,
    const float* __restrict__ W, const float* __restrict__ bias,
    __bf16* __restrict__ Yh, float* __restrict__ Yf) {
  constexpr int K = DIM;
  constexpr int LDT = 40;  // 32 + 8 pad, keeps 16B alignment (80B rows)
  __shared__ __bf16 At[128 * LDT];
  __shared__ __bf16 Bt[128 * LDT];

  const int m0 = blockIdx.y * 128, n0 = blockIdx.x * 128;
  const int t = threadIdx.x;
  const int wave = t >> 5, lane = t & 31, hf = lane >> 4, ln = lane & 15;
  const int wm = wave & 3;   // 4 row strips of 32
  const int wn = wave >> 2;  // 2 col strips of 64

  v8f acc[2][4];
#pragma unroll
  for (int i = 0; i < 2; ++i)
#pragma unroll
    for (int j = 0; j < 4; ++j) acc[i][j] = vzero8();

  const int lrow = t >> 1, lc = (t & 1) * 16;

  for (int k0 = 0; k0 < K; k0 += 32) {
    __syncthreads();
    if (MODE == 0) {
      const float* s = Af + (size_t)(m0 + lrow) * K + k0 + lc;
#pragma unroll
      for (int i = 0; i < 16; ++i) At[lrow * LDT + lc + i] = f2bf(s[i]);
    } else {
      const __bf16* s = Ab + (size_t)(m0 + lrow) * K + k0 + lc;
      const unsigned dst = (unsigned)(uintptr_t)(At + lrow * LDT + lc);
      async_ldg_b128(dst, s);
      async_ldg_b128(dst + 16u, s + 8);
    }
    {
      const float* s = W + (size_t)(n0 + lrow) * K + k0 + lc;
#pragma unroll
      for (int i = 0; i < 16; ++i) Bt[lrow * LDT + lc + i] = f2bf(s[i]);
    }
    if (k0 + 32 < K) {  // speculative prefetch of next W tile -> global_prefetch_b8
      __builtin_prefetch(W + (size_t)(n0 + lrow) * K + k0 + 32 + lc, 0, 0);
    }
    if (MODE == 1) wait_async0();
    __syncthreads();

    v16bf a0 = load_a_frag(At + (wm * 32) * LDT, LDT);
    v16bf a1 = load_a_frag(At + (wm * 32 + 16) * LDT, LDT);
#pragma unroll
    for (int jt = 0; jt < 4; ++jt) {
      v16bf bf = load_b_frag(Bt + (wn * 64 + jt * 16) * LDT, LDT);
      acc[0][jt] = wmma_bf16(a0, bf, acc[0][jt]);
      acc[1][jt] = wmma_bf16(a1, bf, acc[1][jt]);
    }
  }

#pragma unroll
  for (int i = 0; i < 2; ++i)
#pragma unroll
    for (int jt = 0; jt < 4; ++jt)
#pragma unroll
      for (int j = 0; j < 8; ++j) {
        const int m = m0 + wm * 32 + i * 16 + j + hf * 8;
        const int n = n0 + wn * 64 + jt * 16 + ln;
        const float v = acc[i][jt][j] + bias[n];
        if (MODE == 0) {
          const int b = m >> 11, s = m & (SS - 1);
          const int h = n >> 6, d = n & (DEPTH - 1);
          Yh[(((size_t)b * HH + h) * SS + s) * DEPTH + d] = f2bf(v);
        } else {
          Yf[(size_t)m * DIM + n] = v;
        }
      }
}

// ---------- attention ----------

#define LDK 72    // K tile stride (64+8), 144B rows (16B multiple)
#define LDV 136   // V^T / P strip stride (128+8), 272B rows (16B multiple)

__device__ __forceinline__ void compute_scores(const __bf16* Kt, const v16bf qa0,
                                               const v16bf qa1, v8f acc[8]) {
#pragma unroll
  for (int nt = 0; nt < 8; ++nt) {
    v8f a = vzero8();
    a = wmma_bf16(qa0, load_b_frag(Kt + (nt * 16) * LDK, LDK), a);
    a = wmma_bf16(qa1, load_b_frag(Kt + (nt * 16) * LDK + 32, LDK), a);
    acc[nt] = a;
  }
}

__global__ __launch_bounds__(256) void attn_kernel(
    const __bf16* __restrict__ Qh, const __bf16* __restrict__ Kh,
    const __bf16* __restrict__ Vh, float* __restrict__ attn,
    __bf16* __restrict__ ctx) {
  __shared__ __bf16 Kt[128 * LDK];
  __shared__ __bf16 Vt[64 * LDV];        // V transposed: [depth][key]
  __shared__ __bf16 Pst[8][16 * LDV];    // per-wave P strip [16 q rows][128 keys]

  const int qt = blockIdx.x, h = blockIdx.y, b = blockIdx.z;
  const int bh = b * HH + h;
  const int t = threadIdx.x;
  const int wave = t >> 5, lane = t & 31, hf = lane >> 4, ln = lane & 15;
  const int q0 = qt * 128;
  const int r0 = q0 + wave * 16;  // this wave's first q row

  // Q fragments held in registers for both passes (16 rows x 64 depth)
  const __bf16* Qbase = Qh + ((size_t)bh * SS + r0) * DEPTH;
  const v16bf qa0 = load_a_frag(Qbase + 0, DEPTH);
  const v16bf qa1 = load_a_frag(Qbase + 32, DEPTH);

  float mrow[8], srow[8];
#pragma unroll
  for (int j = 0; j < 8; ++j) { mrow[j] = -__builtin_inff(); srow[j] = 0.f; }

  const int lrow = t >> 1, lc = (t & 1) * 32;
  const unsigned KtDst = (unsigned)(uintptr_t)(Kt + lrow * LDK + lc);

  // ---- pass 1: running row max & exp-sum ----
  for (int kt = 0; kt <= qt; ++kt) {
    __syncthreads();
    {
      const __bf16* s = Kh + ((size_t)bh * SS + kt * 128 + lrow) * DEPTH + lc;
#pragma unroll
      for (int i = 0; i < 32; i += 8)
        async_ldg_b128(KtDst + (unsigned)(i * sizeof(__bf16)), s + i);
    }
    wait_async0();
    __syncthreads();

    v8f acc[8];
    compute_scores(Kt, qa0, qa1, acc);

#pragma unroll
    for (int j = 0; j < 8; ++j) {
      const int row = r0 + j + hf * 8;
      float tmax = -__builtin_inff();
#pragma unroll
      for (int nt = 0; nt < 8; ++nt) {
        const int col = kt * 128 + nt * 16 + ln;
        const float v = (col <= row) ? acc[nt][j] * SCALE : -__builtin_inff();
        tmax = fmaxf(tmax, v);
      }
#pragma unroll
      for (int msk = 1; msk < 16; msk <<= 1)
        tmax = fmaxf(tmax, __shfl_xor(tmax, msk, 32));
      const float nm = fmaxf(mrow[j], tmax);
      float ps = 0.f;
#pragma unroll
      for (int nt = 0; nt < 8; ++nt) {
        const int col = kt * 128 + nt * 16 + ln;
        if (col <= row) ps += __expf(acc[nt][j] * SCALE - nm);
      }
#pragma unroll
      for (int msk = 1; msk < 16; msk <<= 1) ps += __shfl_xor(ps, msk, 32);
      srow[j] = srow[j] * __expf(mrow[j] - nm) + ps;
      mrow[j] = nm;
    }
  }

  float inv_s[8];
#pragma unroll
  for (int j = 0; j < 8; ++j) inv_s[j] = 1.f / srow[j];

  // ---- pass 2: write normalized attn, accumulate ctx = P @ V ----
  v8f cacc[4];
#pragma unroll
  for (int j = 0; j < 4; ++j) cacc[j] = vzero8();

  for (int kt = 0; kt < SS / 128; ++kt) {
    if (kt <= qt) {
      __syncthreads();
      {
        const __bf16* ks = Kh + ((size_t)bh * SS + kt * 128 + lrow) * DEPTH + lc;
#pragma unroll
        for (int i = 0; i < 32; i += 8)
          async_ldg_b128(KtDst + (unsigned)(i * sizeof(__bf16)), ks + i);
        const __bf16* vs = Vh + ((size_t)bh * SS + kt * 128 + lrow) * DEPTH + lc;
#pragma unroll
        for (int i = 0; i < 32; ++i)  // transpose into Vt[depth][key]
          Vt[(lc + i) * LDV + lrow] = vs[i];
      }
      wait_async0();
      __syncthreads();

      v8f acc[8];
      compute_scores(Kt, qa0, qa1, acc);

      __bf16* Pw = &Pst[wave][0];
#pragma unroll
      for (int nt = 0; nt < 8; ++nt) {
#pragma unroll
        for (int j = 0; j < 8; ++j) {
          const int row = r0 + j + hf * 8;
          const int col = kt * 128 + nt * 16 + ln;
          const float p =
              (col <= row) ? __expf(acc[nt][j] * SCALE - mrow[j]) * inv_s[j] : 0.f;
          attn[((size_t)bh * SS + row) * SS + col] = p;
          Pw[(j + hf * 8) * LDV + nt * 16 + ln] = f2bf(p);
        }
      }
      // ctx(16x64) += P(16x128) * V(128x64), K-steps of 32 (LDS ops in-order per wave)
#pragma unroll
      for (int kk = 0; kk < 4; ++kk) {
        v16bf pa = load_a_frag(Pw + kk * 32, LDV);
#pragma unroll
        for (int dt = 0; dt < 4; ++dt) {
          v16bf bv = load_b_frag(Vt + (dt * 16) * LDV + kk * 32, LDV);
          cacc[dt] = wmma_bf16(pa, bv, cacc[dt]);
        }
      }
    } else {
      // causal upper region: attn = 0 (wide 16B stores, this is ~250MB of traffic)
      const float4 z4 = make_float4(0.f, 0.f, 0.f, 0.f);
#pragma unroll
      for (int j2 = 0; j2 < 16; ++j2) {
        const int row = r0 + j2;
        *(float4*)(attn + ((size_t)bh * SS + row) * SS + kt * 128 + lane * 4) = z4;
      }
    }
  }

  // merge heads: ctx[b][s][h*64+d], bf16
#pragma unroll
  for (int dt = 0; dt < 4; ++dt)
#pragma unroll
    for (int j = 0; j < 8; ++j) {
      const int row = r0 + j + hf * 8;
      const int d = h * DEPTH + dt * 16 + ln;
      ctx[((size_t)b * SS + row) * DIM + d] = f2bf(cacc[dt][j]);
    }
}

// ---------- launch ----------

extern "C" void kernel_launch(void* const* d_in, const int* in_sizes, int n_in,
                              void* d_out, int out_size, void* d_ws, size_t ws_size,
                              hipStream_t stream) {
  const float* v    = (const float*)d_in[0];
  const float* k    = (const float*)d_in[1];
  const float* q    = (const float*)d_in[2];
  // d_in[3] = mask (causal, implemented analytically)
  const float* wq_w = (const float*)d_in[4];
  const float* wq_b = (const float*)d_in[5];
  const float* wk_w = (const float*)d_in[6];
  const float* wk_b = (const float*)d_in[7];
  const float* wv_w = (const float*)d_in[8];
  const float* wv_b = (const float*)d_in[9];
  const float* dw   = (const float*)d_in[10];
  const float* db   = (const float*)d_in[11];

  float* outp = (float*)d_out;                       // [B,S,D] f32
  float* attn = outp + (size_t)BB * SS * DIM;        // [B,H,S,S] f32

  const size_t NE = (size_t)BB * SS * DIM;           // 4,194,304 elements
  __bf16* Qh = (__bf16*)d_ws;                        // [B,H,S,DEPTH]
  __bf16* Kh = Qh + NE;
  __bf16* Vh = Kh + NE;
  __bf16* Cx = Vh + NE;                              // [B,S,D] bf16 (32 MB total ws)

  dim3 gg(DIM / 128, (BB * SS) / 128);               // (8, 32)
  gemm_kernel<0><<<gg, 256, 0, stream>>>(q, nullptr, wq_w, wq_b, Qh, nullptr);
  gemm_kernel<0><<<gg, 256, 0, stream>>>(k, nullptr, wk_w, wk_b, Kh, nullptr);
  gemm_kernel<0><<<gg, 256, 0, stream>>>(v, nullptr, wv_w, wv_b, Vh, nullptr);

  attn_kernel<<<dim3(SS / 128, HH, BB), 256, 0, stream>>>(Qh, Kh, Vh, attn, Cx);

  gemm_kernel<1><<<gg, 256, 0, stream>>>(nullptr, Cx, dw, db, nullptr, outp);
}
